// MultiHeadAttention_10582799418017
// MI455X (gfx1250) — matrix operand order repaired and checked by reading, not compiled, not run on hardware
//
#include <hip/hip_runtime.h>
#include <hip/hip_bf16.h>

// MHA forward for gfx1250 (MI455X): bf16 WMMA flash-attention with
// async global->LDS staging of K/V tiles (shared by all 8 waves of a block).
// B=2, S=2048, E=512, H=8, D=64.
#define BATCH 2
#define SEQ   2048
#define EMBED 512
#define NH    8
#define DH    64

typedef __attribute__((ext_vector_type(16))) __bf16 v16bf;
typedef __attribute__((ext_vector_type(2)))  __bf16 v2bf;
typedef __attribute__((ext_vector_type(8)))  float  v8f;
typedef __attribute__((ext_vector_type(2)))  float  v2f;

union BF16Frag {
  v16bf v;
  unsigned short h[16];
  unsigned int   u[8];
};

// Native f32->bf16 conversions (v_cvt_pk_bf16_f32 class ops).
__device__ __forceinline__ unsigned short f2bf(float f) {
  __bf16 b = (__bf16)f;
  return __builtin_bit_cast(unsigned short, b);
}
__device__ __forceinline__ unsigned int pack2bf(float a, float b) {
  v2bf r = __builtin_convertvector((v2f){a, b}, v2bf);
  return __builtin_bit_cast(unsigned int, r);
}

// ---------------------------------------------------------------- weights -> bf16
__global__ void wconv_kernel(const float* __restrict__ src,
                             unsigned short* __restrict__ dst, int n) {
  int i = blockIdx.x * blockDim.x + threadIdx.x;
  if (i < n) dst[i] = f2bf(src[i]);
}

// ---------------------------------------------------------------- Q/K/V projection
// rows r = (b*H + h)*S + s ; out[r][e] = sum_d X[b,s,h*64+d] * W[e][d]
// One wave: 16 rows x 64 cols, K=64 (2 WMMA k-steps), 8 WMMAs total.
__global__ __launch_bounds__(256) void qkv_proj_kernel(
    const float* __restrict__ X,              // [B,S,512] fp32
    const unsigned short* __restrict__ Wbf,   // [64,64] bf16 (torch [out,in])
    unsigned short* __restrict__ Out,         // Q/K: [B*H*S,64] ; V: [B*H*64,S]
    int transposeOut) {
  const int lane = threadIdx.x & 31;
  const int wave = threadIdx.x >> 5;
  const int nlo  = lane & 15;
  const int hi   = lane >> 4;
  const int tile = blockIdx.x * 8 + wave;     // 16-row tile index
  const int r0   = tile * 16;

  // A fragment rows: m = nlo -> row r0+nlo -> (b,h,s)
  const int r  = r0 + nlo;
  const int s  = r & (SEQ - 1);
  const int bh = r >> 11;                     // r / SEQ
  const int h  = bh & (NH - 1);
  const int b  = bh >> 3;
  const float* xrow = X + ((size_t)(b * SEQ + s) * EMBED) + h * DH;

  BF16Frag a[2];
#pragma unroll
  for (int t = 0; t < 2; ++t)
#pragma unroll
    for (int p = 0; p < 8; ++p) {
      int dw = p + 8 * hi;  // A-layout k-map (pairs)
      int d  = t * 32 + dw * 2;
      a[t].u[p] = pack2bf(xrow[d], xrow[d + 1]);
    }

  v8f acc[4];
#pragma unroll
  for (int j = 0; j < 4; ++j) {
    int e = j * 16 + nlo;                     // B fragment: lane n = output feature
    v8f c = (v8f){0.f,0.f,0.f,0.f,0.f,0.f,0.f,0.f};
#pragma unroll
    for (int t = 0; t < 2; ++t) {
      BF16Frag bfr;
#pragma unroll
      for (int p = 0; p < 8; ++p) {
        int k = t * 32 + 2 * p + 16 * hi;     // B-layout k-map
        bfr.u[p] = *(const unsigned int*)&Wbf[e * DH + k];
      }
      c = __builtin_amdgcn_wmma_f32_16x16x32_bf16(false, a[t].v, false, bfr.v,
                                                  (short)0, c, false, false);
    }
    acc[j] = c;
  }

  if (!transposeOut) {                        // Q/K: row-major [r][d]
#pragma unroll
    for (int j = 0; j < 4; ++j) {
      int col = j * 16 + nlo;
#pragma unroll
      for (int rr = 0; rr < 8; ++rr) {
        int row = r0 + rr + 8 * hi;
        Out[(size_t)row * DH + col] = f2bf(acc[j][rr]);
      }
    }
  } else {                                    // V: transposed [bh*64+d][s], contiguous s per lane
    int s0  = r0 & (SEQ - 1);
    int bh0 = r0 >> 11;
#pragma unroll
    for (int j = 0; j < 4; ++j) {
      int d = j * 16 + nlo;
      unsigned int* dst =
          (unsigned int*)(Out + ((size_t)(bh0 * DH + d)) * SEQ + s0 + 8 * hi);
#pragma unroll
      for (int q = 0; q < 4; ++q)
        dst[q] = pack2bf(acc[j][2 * q], acc[j][2 * q + 1]);
    }
  }
}

// ---------------------------------------------------------------- flash attention
// Block = 8 waves, all on the same (b,h); wave w handles Sq tile sq0..sq0+15.
// Per 32-wide Sk step: the block async-copies K tile (32x64, 4KB) and V^T tile
// (64x32, 4KB) into LDS (one global_load_async_to_lds_b128 per thread per tile),
// then each wave runs 4 WMMAs QK^T + online softmax + LDS P-transpose + 4 WMMAs P*V.
__global__ __launch_bounds__(256) void attn_kernel(
    const unsigned short* __restrict__ Qp,    // [B*H*S,64] bf16
    const unsigned short* __restrict__ Kp,    // [B*H*S,64] bf16
    const unsigned short* __restrict__ Vpt,   // [B*H*64,S] bf16 (V^T per head)
    const int* __restrict__ mask,             // [B,1,S,S] int32
    unsigned short* __restrict__ Ctx)         // [B*S,512] bf16
{
  __shared__ unsigned short ldsK[32 * DH];    // K rows sk0..sk0+31, row-major [32][64]
  __shared__ unsigned short ldsV[DH * 32];    // V^T rows d=0..63, row-major [64][32]
  __shared__ unsigned short ldsP[8][16][32];  // per-wave 16x32 P tile (1KB/wave)

  const int tid  = threadIdx.x;
  const int lane = tid & 31;
  const int wave = tid >> 5;
  const int nlo  = lane & 15;
  const int hi   = lane >> 4;
  const int w    = blockIdx.x * 8 + wave;
  const int sq0  = (w & 127) * 16;
  const int bh   = (blockIdx.x * 8) >> 7;     // uniform across the block
  const int b    = bh >> 3;
  const int h    = bh & 7;

  // ---- staging addresses (one b128 per thread per tile)
  const unsigned long long kbase =
      (unsigned long long)(size_t)(Kp + (size_t)(bh * SEQ) * DH);
  const unsigned long long vbase =
      (unsigned long long)(size_t)(Vpt + (size_t)(bh * DH) * SEQ);
  const unsigned kLds = ((unsigned)(size_t)&ldsK[0]) + tid * 16;
  const unsigned vLds = ((unsigned)(size_t)&ldsV[0]) + (tid >> 2) * 64 + (tid & 3) * 16;
  const unsigned kGofBase = (unsigned)tid * 16;                        // contiguous 4KB
  const unsigned vGofBase = (((tid >> 2) * SEQ) + (tid & 3) * 8) * 2;  // 64B per d-row

  // persistent Q A-fragments (K=64 -> 2 k-steps)
  BF16Frag qf[2];
  {
    const unsigned short* qrow = Qp + ((size_t)(bh * SEQ) + sq0 + nlo) * DH;
#pragma unroll
    for (int t = 0; t < 2; ++t)
#pragma unroll
      for (int p = 0; p < 8; ++p) {
        int dw = p + 8 * hi;
        qf[t].u[p] = *(const unsigned int*)&qrow[t * 32 + dw * 2];
      }
  }

  float mrow[8], lrow[8];
  v8f o[4];
#pragma unroll
  for (int rr = 0; rr < 8; ++rr) { mrow[rr] = -3.0e38f; lrow[rr] = 0.f; }
#pragma unroll
  for (int j = 0; j < 4; ++j) o[j] = (v8f){0,0,0,0,0,0,0,0};

  const int* mbase = mask + (size_t)b * SEQ * SEQ;
  const unsigned int* ldsK32 = (const unsigned int*)&ldsK[0];
  const unsigned int* ldsV32 = (const unsigned int*)&ldsV[0];

  for (int sk0 = 0; sk0 < SEQ; sk0 += 32) {
    // ---- async stage K (4KB) and V^T (4KB) tiles into LDS
    {
      unsigned kGof = kGofBase + (unsigned)sk0 * (DH * 2);
      unsigned vGof = vGofBase + (unsigned)sk0 * 2;
      asm volatile("global_load_async_to_lds_b128 %0, %1, %2"
                   :: "v"(kLds), "v"(kGof), "s"(kbase) : "memory");
      asm volatile("global_load_async_to_lds_b128 %0, %1, %2"
                   :: "v"(vLds), "v"(vGof), "s"(vbase) : "memory");
      asm volatile("s_wait_asynccnt 0x0" ::: "memory");
    }
    __syncthreads();                          // tiles visible to all waves

    // ---- scores for two 16-col sub-tiles (K fragments from LDS)
    v8f sc[2];
#pragma unroll
    for (int u = 0; u < 2; ++u) {
      int skl = u * 16 + nlo;                 // row within the 32-row K tile
      v8f c = (v8f){0,0,0,0,0,0,0,0};
#pragma unroll
      for (int t = 0; t < 2; ++t) {
        BF16Frag kf;
#pragma unroll
        for (int p = 0; p < 8; ++p)
          kf.u[p] = ldsK32[skl * 32 + t * 16 + p + 8 * hi];
        c = __builtin_amdgcn_wmma_f32_16x16x32_bf16(false, qf[t].v, false, kf.v,
                                                    (short)0, c, false, false);
      }
      // mask BEFORE scale (faithful to reference)
      int skc = sk0 + skl;
#pragma unroll
      for (int rr = 0; rr < 8; ++rr) {
        int sq = sq0 + rr + 8 * hi;
        int mk = mbase[(size_t)sq * SEQ + skc];
        float v = (mk != 0) ? c[rr] : -1.0e20f;
        c[rr] = v * 0.125f;                   // 1/sqrt(64)
      }
      sc[u] = c;
    }

    // ---- online softmax state update (butterfly over the 16 n-lanes)
    float mnew[8], csc[8];
#pragma unroll
    for (int rr = 0; rr < 8; ++rr) {
      float t = fmaxf(sc[0][rr], sc[1][rr]);
#pragma unroll
      for (int x = 1; x < 16; x <<= 1) t = fmaxf(t, __shfl_xor(t, x, 32));
      mnew[rr] = fmaxf(mrow[rr], t);
      csc[rr]  = __expf(mrow[rr] - mnew[rr]);
      mrow[rr] = mnew[rr];
    }
#pragma unroll
    for (int rr = 0; rr < 8; ++rr) {
      float p0 = __expf(sc[0][rr] - mnew[rr]);
      float p1 = __expf(sc[1][rr] - mnew[rr]);
      sc[0][rr] = p0; sc[1][rr] = p1;
      float t = p0 + p1;
#pragma unroll
      for (int x = 1; x < 16; x <<= 1) t += __shfl_xor(t, x, 32);
      lrow[rr] = lrow[rr] * csc[rr] + t;
    }
#pragma unroll
    for (int j = 0; j < 4; ++j)
#pragma unroll
      for (int rr = 0; rr < 8; ++rr) o[j][rr] *= csc[rr];

    // ---- transpose P (C-layout -> A-layout) through wave-private LDS
#pragma unroll
    for (int rr = 0; rr < 8; ++rr) {
      int row = rr + 8 * hi;
      ldsP[wave][row][nlo]      = f2bf(sc[0][rr]);
      ldsP[wave][row][16 + nlo] = f2bf(sc[1][rr]);
    }
    asm volatile("s_wait_dscnt 0x0" ::: "memory");  // cross-lane RAW in LDS
    BF16Frag pf;
    {
      const unsigned short* prow = &ldsP[wave][nlo][0];
#pragma unroll
      for (int p = 0; p < 8; ++p) {
        int dw = p + 8 * hi;
        pf.u[p] = *(const unsigned int*)&prow[dw * 2];
      }
    }

    // ---- O += P x V  (4 d-tiles, K=32; V fragments from LDS)
#pragma unroll
    for (int j = 0; j < 4; ++j) {
      int d = j * 16 + nlo;
      BF16Frag vf;
#pragma unroll
      for (int p = 0; p < 8; ++p)
        vf.u[p] = ldsV32[d * 16 + p + 8 * hi];
      o[j] = __builtin_amdgcn_wmma_f32_16x16x32_bf16(false, pf.v, false, vf.v,
                                                     (short)0, o[j], false, false);
    }
    __syncthreads();                          // all waves done before next overwrite
  }

  // ---- normalize and emit ctx (row-major bf16 [B*S, 512])
  float inv[8];
#pragma unroll
  for (int rr = 0; rr < 8; ++rr) inv[rr] = 1.0f / lrow[rr];
#pragma unroll
  for (int j = 0; j < 4; ++j) {
    int col = h * DH + j * 16 + nlo;
#pragma unroll
    for (int rr = 0; rr < 8; ++rr) {
      int sq = sq0 + rr + 8 * hi;
      Ctx[(size_t)(b * SEQ + sq) * EMBED + col] = f2bf(o[j][rr] * inv[rr]);
    }
  }
}

// ---------------------------------------------------------------- output projection
// out = ctx @ Wo^T + bo ; M=4096, N=512, K=512. One wave: 16x64 tile, 64 WMMAs.
__global__ __launch_bounds__(256) void out_proj_kernel(
    const unsigned short* __restrict__ Ctx,   // [B*S,512] bf16
    const unsigned short* __restrict__ Wob,   // [512,512] bf16
    const float* __restrict__ bo,             // [512] f32
    float* __restrict__ Out)                  // [B*S,512] f32
{
  const int lane = threadIdx.x & 31;
  const int wave = threadIdx.x >> 5;
  const int nlo  = lane & 15;
  const int hi   = lane >> 4;
  const int w    = blockIdx.x * 8 + wave;
  const int e0   = (w & 7) * 64;
  const int r0   = (w >> 3) * 16;

  v8f acc[4];
#pragma unroll
  for (int j = 0; j < 4; ++j) acc[j] = (v8f){0,0,0,0,0,0,0,0};

  const unsigned short* arow = Ctx + (size_t)(r0 + nlo) * EMBED;
  for (int t = 0; t < 16; ++t) {
    BF16Frag af;
#pragma unroll
    for (int p = 0; p < 8; ++p) {
      int dw = p + 8 * hi;
      af.u[p] = *(const unsigned int*)&arow[t * 32 + dw * 2];
    }
#pragma unroll
    for (int j = 0; j < 4; ++j) {
      int e = e0 + j * 16 + nlo;
      const unsigned short* wrow = Wob + (size_t)e * EMBED + t * 32;
      BF16Frag bfr;
#pragma unroll
      for (int p = 0; p < 8; ++p)
        bfr.u[p] = *(const unsigned int*)&wrow[2 * p + 16 * hi];
      acc[j] = __builtin_amdgcn_wmma_f32_16x16x32_bf16(false, af.v, false, bfr.v,
                                                       (short)0, acc[j], false, false);
    }
  }
#pragma unroll
  for (int j = 0; j < 4; ++j) {
    int e = e0 + j * 16 + nlo;
    float bias = bo[e];
#pragma unroll
    for (int rr = 0; rr < 8; ++rr) {
      int row = r0 + rr + 8 * hi;
      Out[(size_t)row * EMBED + e] = acc[j][rr] + bias;
    }
  }
}

// ---------------------------------------------------------------- launch
extern "C" void kernel_launch(void* const* d_in, const int* in_sizes, int n_in,
                              void* d_out, int out_size, void* d_ws, size_t ws_size,
                              hipStream_t stream) {
  const float* key   = (const float*)d_in[0];
  const float* query = (const float*)d_in[1];
  const float* value = (const float*)d_in[2];
  const int*   mask  = (const int*)d_in[3];
  const float* Wq    = (const float*)d_in[4];
  const float* Wk    = (const float*)d_in[5];
  const float* Wv    = (const float*)d_in[6];
  const float* Wo    = (const float*)d_in[7];
  const float* bo    = (const float*)d_in[8];
  float* out = (float*)d_out;

  char* ws = (char*)d_ws;
  size_t off = 0;
  const size_t szHead = (size_t)BATCH * NH * SEQ * DH * sizeof(unsigned short); // 4 MiB
  unsigned short* Qp  = (unsigned short*)(ws + off); off += szHead;
  unsigned short* Kp  = (unsigned short*)(ws + off); off += szHead;
  unsigned short* Vpt = (unsigned short*)(ws + off); off += szHead;
  unsigned short* Ctx = (unsigned short*)(ws + off); off += (size_t)BATCH * SEQ * EMBED * 2;
  unsigned short* Wqb = (unsigned short*)(ws + off); off += DH * DH * 2;
  unsigned short* Wkb = (unsigned short*)(ws + off); off += DH * DH * 2;
  unsigned short* Wvb = (unsigned short*)(ws + off); off += DH * DH * 2;
  unsigned short* Wob = (unsigned short*)(ws + off); off += (size_t)EMBED * EMBED * 2;

  // 1) weights -> bf16
  wconv_kernel<<<(DH * DH + 255) / 256, 256, 0, stream>>>(Wq, Wqb, DH * DH);
  wconv_kernel<<<(DH * DH + 255) / 256, 256, 0, stream>>>(Wk, Wkb, DH * DH);
  wconv_kernel<<<(DH * DH + 255) / 256, 256, 0, stream>>>(Wv, Wvb, DH * DH);
  wconv_kernel<<<(EMBED * EMBED + 255) / 256, 256, 0, stream>>>(Wo, Wob, EMBED * EMBED);

  // 2) head projections (2048 row-tiles / 8 waves per block = 256 blocks)
  qkv_proj_kernel<<<256, 256, 0, stream>>>(query, Wqb, Qp, 0);
  qkv_proj_kernel<<<256, 256, 0, stream>>>(key,   Wkb, Kp, 0);
  qkv_proj_kernel<<<256, 256, 0, stream>>>(value, Wvb, Vpt, 1);

  // 3) flash attention: B*H*(S/16) = 2048 waves -> 256 blocks (8 waves share K/V tiles)
  attn_kernel<<<256, 256, 0, stream>>>(Qp, Kp, Vpt, mask, Ctx);

  // 4) output projection: 2048 waves -> 256 blocks
  out_proj_kernel<<<256, 256, 0, stream>>>(Ctx, Wob, bo, out);
}
